// CachedMLP_25177098289318
// MI455X (gfx1250) — matrix operand order repaired and checked by reading, not compiled, not run on hardware
//
#include <hip/hip_runtime.h>

// ---------------- types ----------------
typedef __attribute__((ext_vector_type(16))) _Float16 v16h;
typedef __attribute__((ext_vector_type(8)))  _Float16 v8h;
typedef __attribute__((ext_vector_type(8)))  float    v8f;
typedef __attribute__((ext_vector_type(4)))  float    f4;

#define D_MODEL 4096
#define HIDDEN  14336
#define ACTIVE  11468
#define TOKENS  32
#define JPAD    11520      // ACTIVE padded to 360*32
#define NBLK    180        // j-blocks of 64 per expert
#define KSEG    8
#define KC_TOT  720        // 2*JPAD/32
#define KC_PER_SEG 90      // KC_TOT / KSEG

// B-operand fragment: 16 consecutive f32 at p, streamed non-temporally,
// converted to fp16 (contiguous k order, same as A: lane=n, halves split K by 16).
static __device__ inline v16h load16h_nt(const float* p) {
  const f4* q = (const f4*)p;
  f4 x[4] = {__builtin_nontemporal_load(q + 0), __builtin_nontemporal_load(q + 1),
             __builtin_nontemporal_load(q + 2), __builtin_nontemporal_load(q + 3)};
  v16h r;
#pragma unroll
  for (int v = 0; v < 4; ++v)
#pragma unroll
    for (int i = 0; i < 4; ++i) r[4 * v + i] = (_Float16)x[v][i];
  return r;
}

// A-operand fragment from an fp16 row: a[0..7] = row[kb+16*half ..],
// a[8..15] = row[kb+16*half+8 ..]  (two 16-byte loads, no conversion)
static __device__ inline v16h loadA16(const _Float16* rowbase, int half) {
  v8h lo = *(const v8h*)(rowbase + 16 * half);
  v8h hi = *(const v8h*)(rowbase + 16 * half + 8);
  v16h r;
#pragma unroll
  for (int i = 0; i < 8; ++i) { r[i] = lo[i]; r[8 + i] = hi[i]; }
  return r;
}

static __device__ inline v8f wmma_f16(v16h a, v16h b, v8f c) {
  // (neg_a, A, neg_b, B, c_mod, C, reuse_a, reuse_b)
  return __builtin_amdgcn_wmma_f32_16x16x32_f16(false, a, false, b, (short)0, c,
                                                false, false);
}

// ---------------------------------------------------------------------------
// Kernel 0: h (f32) -> h16 (fp16), once per launch
// ---------------------------------------------------------------------------
__global__ __launch_bounds__(256) void cvt_h(const float* __restrict__ h,
                                             _Float16* __restrict__ h16) {
  const int i = blockIdx.x * 256 + threadIdx.x;
  if (i < TOKENS * D_MODEL) h16[i] = (_Float16)h[i];
}

// ---------------------------------------------------------------------------
// Kernel 1: A16[e][t][j] = ew_e * silu(h·w1_e[j]) * (h·w3_e[idx0[j]])
// grid.x = 2*NBLK, block = 256 (8 waves; wave owns one M-tile x one 16-wide j-tile)
// ---------------------------------------------------------------------------
__global__ __launch_bounds__(256) void moe_gateup(
    const _Float16* __restrict__ h16,
    const float* __restrict__ w3_0, const float* __restrict__ w3_1,
    const float* __restrict__ w1_0, const float* __restrict__ w1_1,
    const float* __restrict__ ew, const int* __restrict__ ids,
    const int* __restrict__ idx0,
    _Float16* __restrict__ A) {
  const int e    = blockIdx.x / NBLK;
  const int jblk = blockIdx.x % NBLK;
  const int wid  = threadIdx.x >> 5;
  const int lane = threadIdx.x & 31;
  const int mt   = wid & 1;        // which 16-token M-tile
  const int nt   = wid >> 1;       // which 16-wide j-tile (of 4)
  const int nl   = lane & 15;
  const int half = lane >> 4;

  const int n  = jblk * 64 + nt * 16 + nl;        // this lane's j column
  const int nc = (n < ACTIVE) ? n : (ACTIVE - 1); // clamp for padded tail
  const int gi = idx0[nc];                        // gathered w3 row

  const float* w1 = e ? w1_1 : w1_0;
  const float* w3 = e ? w3_1 : w3_0;

  const bool  swap = (ids[0] != 0);
  const float ewe  = ew[swap ? (1 - e) : e];

  const float*    p1 = w1 + (size_t)nc * D_MODEL + half * 16;
  const float*    p3 = w3 + (size_t)gi * D_MODEL + half * 16;
  const _Float16* pa = h16 + (size_t)(nl + 16 * mt) * D_MODEL;

  v8f c1 = {}, c3 = {};

  for (int kb = 0; kb < D_MODEL; kb += 32) {
    v16h b1 = load16h_nt(p1 + kb);
    v16h b3 = load16h_nt(p3 + kb);
    v16h a  = loadA16(pa + kb, half);
    c1 = wmma_f16(a, b1, c1);
    c3 = wmma_f16(a, b3, c3);
  }

  _Float16* Ae = A + (size_t)e * TOKENS * JPAD;
  const bool valid = (n < ACTIVE);
#pragma unroll
  for (int r = 0; r < 8; ++r) {
    // C layout: VGPR r, lanes 0-15 -> M=r, lanes 16-31 -> M=r+8; N = lane&15
    const int t = r + 8 * half + 16 * mt;
    float g = c1[r], u = c3[r];
    float v = valid ? (ewe * u * (g / (1.0f + __expf(-g)))) : 0.0f;
    Ae[(size_t)t * JPAD + n] = (_Float16)v;
  }
}

// ---------------------------------------------------------------------------
// Kernel 2: P[s][t][d] = sum over K-segment s of A[e][t][j] * w2_e[j][d]
// grid.x = 64 d-blocks * KSEG, block = 256 (8 waves; wave owns M-tile x 16 d-cols)
// ---------------------------------------------------------------------------
__global__ __launch_bounds__(256) void moe_down(
    const _Float16* __restrict__ A,
    const float* __restrict__ w2_0, const float* __restrict__ w2_1,
    float* __restrict__ P) {
  const int dblk = blockIdx.x & 63;
  const int kseg = blockIdx.x >> 6;
  const int wid  = threadIdx.x >> 5;
  const int lane = threadIdx.x & 31;
  const int mt   = wid & 1;
  const int nt   = wid >> 1;
  const int nl   = lane & 15;
  const int half = lane >> 4;
  const int d    = dblk * 64 + nt * 16 + nl;

  v8f c = {};

  for (int kc = 0; kc < KC_PER_SEG; ++kc) {
    const int gkc = kseg * KC_PER_SEG + kc;
    const int e   = (gkc >= KC_TOT / 2) ? 1 : 0;   // expert uniform per K-chunk
    const int jb  = (gkc - e * (KC_TOT / 2)) * 32;

    const _Float16* Ae = A + (size_t)e * TOKENS * JPAD;
    const float*    w2 = e ? w2_1 : w2_0;

    v16h a = loadA16(Ae + (size_t)(nl + 16 * mt) * JPAD + jb, half);

    // B-operand: b[i] = w2[jb + 16*half + i][d]; rows past ACTIVE clamp (A pad=0)
    v16h b;
#pragma unroll
    for (int i = 0; i < 16; ++i) {
      int j = jb + 16 * half + i;
      j = (j < ACTIVE) ? j : (ACTIVE - 1);
      b[i] = (_Float16)__builtin_nontemporal_load(&w2[(size_t)j * D_MODEL + d]);
    }

    c = wmma_f16(a, b, c);
  }

  float* Ps = P + (size_t)kseg * TOKENS * D_MODEL;
#pragma unroll
  for (int r = 0; r < 8; ++r)
    Ps[(size_t)(r + 8 * half + 16 * mt) * D_MODEL + d] = c[r];
}

// ---------------------------------------------------------------------------
// Kernel 3: out[t][d] = sum_s P[s][t][d]
// ---------------------------------------------------------------------------
__global__ __launch_bounds__(256) void moe_reduce(const float* __restrict__ P,
                                                  float* __restrict__ out) {
  const int i = blockIdx.x * 256 + threadIdx.x;
  if (i < TOKENS * D_MODEL) {
    float s = 0.0f;
#pragma unroll
    for (int k = 0; k < KSEG; ++k) s += P[(size_t)k * TOKENS * D_MODEL + i];
    out[i] = s;
  }
}

// ---------------------------------------------------------------------------
extern "C" void kernel_launch(void* const* d_in, const int* in_sizes, int n_in,
                              void* d_out, int out_size, void* d_ws, size_t ws_size,
                              hipStream_t stream) {
  (void)in_sizes; (void)n_in; (void)out_size; (void)ws_size;
  const float* h    = (const float*)d_in[0];
  const float* w3_0 = (const float*)d_in[1];
  const float* w3_1 = (const float*)d_in[2];
  const float* w1_0 = (const float*)d_in[3];
  const float* w2_0 = (const float*)d_in[4];
  const float* w1_1 = (const float*)d_in[5];
  const float* w2_1 = (const float*)d_in[6];
  const float* ew   = (const float*)d_in[7];
  const int*   idx0 = (const int*)d_in[8];
  const int*   ids  = (const int*)d_in[9];

  char* base = (char*)d_ws;
  _Float16* h16 = (_Float16*)base;                         // 256 KB
  _Float16* A16 = (_Float16*)(base + 262144);              // [2][32][JPAD] 1.44 MB
  float*    P   = (float*)(base + 262144 + 1474560);       // [KSEG][32][4096] 4 MB

  cvt_h<<<(TOKENS * D_MODEL + 255) / 256, 256, 0, stream>>>(h, h16);
  moe_gateup<<<2 * NBLK, 256, 0, stream>>>(h16, w3_0, w3_1, w1_0, w1_1, ew, ids,
                                           idx0, A16);
  moe_down<<<64 * KSEG, 256, 0, stream>>>(A16, w2_0, w2_1, P);
  moe_reduce<<<(TOKENS * D_MODEL + 255) / 256, 256, 0, stream>>>(P,
                                                                 (float*)d_out);
}
